// HierarchicalMILModel_18932215841361
// MI455X (gfx1250) — compile-verified, matrix operand order checked by reading, not run on hardware
//
#include <hip/hip_runtime.h>
#include <hip/hip_bf16.h>
#include <math.h>

typedef __attribute__((ext_vector_type(16))) _Float16 v16h;
typedef __attribute__((ext_vector_type(8)))  float    v8f;
typedef unsigned int u32;

#define NCL   11            // C+1
#define IDIM  512
#define BB    8
#define NN    16384
#define TOK   (BB*NN)

__device__ __forceinline__ int fkey(float f) {       // monotonic float->int key (involution)
    int i = __float_as_int(f);
    return i ^ ((i >> 31) & 0x7fffffff);
}
__device__ __forceinline__ float fdec(int m) {
    return __int_as_float(m ^ ((m >> 31) & 0x7fffffff));
}
__device__ __forceinline__ u32 pkh(float a, float b) {
    union { _Float16 h[2]; u32 u; } p;
    p.h[0] = (_Float16)a; p.h[1] = (_Float16)b;
    return p.u;
}

// ---- pack f32 [Kreal x Nn] row-major weights into f16 WMMA B-fragments ----
// fragment f = kt*NT + nt, 1024B: lane(32) x 8 dwords, dword j = (k, k+1)
__global__ void k_pack_b(const float* __restrict__ W, int Kreal, int Nn, int NT,
                         u32* __restrict__ dst, int total) {
    int idx = blockIdx.x * 256 + threadIdx.x;
    if (idx >= total) return;
    int f = idx >> 8, lane = (idx >> 3) & 31, j = idx & 7;
    int kt = f / NT, nt = f % NT;
    int n  = nt * 16 + (lane & 15);
    int hi = lane >> 4;
    int k  = kt * 32 + ((j >> 2) << 4) + hi * 8 + ((j & 3) << 1);
    float a = (k     < Kreal) ? W[(size_t)k * Nn + n]       : 0.0f;
    float b = (k + 1 < Kreal) ? W[(size_t)(k + 1) * Nn + n] : 0.0f;
    dst[(size_t)f * 256 + lane * 8 + j] = pkh(a, b);
}

__global__ void k_init(int* __restrict__ segmax, float* __restrict__ segsum,
                       float* __restrict__ segcnt, float* __restrict__ cfeat) {
    int idx = blockIdx.x * 256 + threadIdx.x;
    if (idx < BB * NCL * 512) cfeat[idx] = 0.0f;
    if (idx < BB * NCL) {
        segmax[idx] = fkey(-__builtin_inff());
        segsum[idx] = 0.0f;
        segcnt[idx] = 0.0f;
    }
}

// ---- heavy kernel: fusion GEMM + gated attention scores, 32 tokens/block ----
__global__ __launch_bounds__(256)
void k_token(const float* __restrict__ h, const int* __restrict__ cid,
             const float* __restrict__ emb,
             const u32* __restrict__ wfp, const float* __restrict__ bfuse,
             const u32* __restrict__ wap, const float* __restrict__ ba1,
             const u32* __restrict__ wbp, const float* __restrict__ bb1,
             const float* __restrict__ wc1, const float* __restrict__ bc1,
             _Float16* __restrict__ xout, float* __restrict__ Aout) {
    __shared__ __align__(16) _Float16 sX[32][512];    // x = relu(fuse), f16
    __shared__ __align__(16) _Float16 sEMB[32][32];   // embedding k-tail (k=512..543)
    __shared__ float sArow[32];                       // per-row attention logit acc

    const int t    = threadIdx.x;
    const int m0   = blockIdx.x * 32;
    const int wv   = t >> 5, lane = t & 31;
    const int hi   = lane >> 4, lrow = lane & 15;
    const int rt   = wv >> 2;                 // row-tile 0/1 (16 rows each)
    const int cs   = wv & 3;                  // column strip
    const int arow = m0 + rt * 16 + lrow;     // global token row for A-fragments

    // stage 0: embedding tail + zero pad + logit-acc init
    if (t < 32) {
        int id = cid[m0 + t];
        id = id < 0 ? 0 : (id > 10 ? 10 : id);
#pragma unroll
        for (int e = 0; e < 8; ++e) sEMB[t][e] = (_Float16)emb[id * 8 + e];
#pragma unroll
        for (int e = 8; e < 32; ++e) sEMB[t][e] = (_Float16)0.0f;
        sArow[t] = bc1[0];
    }
    __syncthreads();

    // stage 1: x = relu([h,emb] @ W_fuse + b) ; A-frags straight from global h
    {
        v8f acc[8];
#pragma unroll
        for (int i = 0; i < 8; ++i) acc[i] = (v8f){0,0,0,0,0,0,0,0};
        const float* hrow = h + (size_t)arow * IDIM;
        for (int kt = 0; kt < 16; ++kt) {             // even trip: clean pipelining
            const float4 fa = *(const float4*)(hrow + kt * 32 + hi * 8);
            const float4 fb = *(const float4*)(hrow + kt * 32 + hi * 8 + 4);
            const float4 fc = *(const float4*)(hrow + kt * 32 + hi * 8 + 16);
            const float4 fd = *(const float4*)(hrow + kt * 32 + hi * 8 + 20);
            union { v16h v; u32 u[8]; } af;
            af.u[0] = pkh(fa.x, fa.y); af.u[1] = pkh(fa.z, fa.w);
            af.u[2] = pkh(fb.x, fb.y); af.u[3] = pkh(fb.z, fb.w);
            af.u[4] = pkh(fc.x, fc.y); af.u[5] = pkh(fc.z, fc.w);
            af.u[6] = pkh(fd.x, fd.y); af.u[7] = pkh(fd.z, fd.w);
#pragma unroll
            for (int i = 0; i < 8; ++i) {
                v16h bf = *(const v16h*)(wfp + ((size_t)(kt * 32 + cs * 8 + i)) * 256 + lane * 8);
                acc[i] = __builtin_amdgcn_wmma_f32_16x16x32_f16(
                    false, af.v, false, bf, (short)0, acc[i], false, false);
            }
        }
        {   // k-tile 16: embedding tail from LDS
            union { v16h v; u32 u[8]; } af;
            const u32* ep = (const u32*)(&sEMB[rt * 16 + lrow][0]);
#pragma unroll
            for (int j = 0; j < 8; ++j) {
                int k = ((j >> 2) << 4) + hi * 8 + ((j & 3) << 1);
                af.u[j] = ep[k >> 1];
            }
#pragma unroll
            for (int i = 0; i < 8; ++i) {
                v16h bf = *(const v16h*)(wfp + ((size_t)(16 * 32 + cs * 8 + i)) * 256 + lane * 8);
                acc[i] = __builtin_amdgcn_wmma_f32_16x16x32_f16(
                    false, af.v, false, bf, (short)0, acc[i], false, false);
            }
        }
#pragma unroll
        for (int i = 0; i < 8; ++i) {
            int col = (cs * 8 + i) * 16 + lrow;
            float bias = bfuse[col];
#pragma unroll
            for (int r = 0; r < 8; ++r) {
                int m = rt * 16 + hi * 8 + r;
                float vl = acc[i][r] + bias;
                sX[m][col] = (_Float16)(vl > 0.0f ? vl : 0.0f);
            }
        }
    }
    __syncthreads();

    // stream x (f16) to global for the cfeat pass: 32x512x2B = 2048 uint4
    {
        const uint4* src = (const uint4*)(&sX[0][0]);
        uint4* dst = (uint4*)(xout + (size_t)m0 * 512);
#pragma unroll
        for (int it = 0; it < 8; ++it) dst[t + it * 256] = src[t + it * 256];
    }

    // stage 2: a = tanh(x@Wa1+ba1), g = sigmoid(x@Wb1+bb1), logit = sum a*g*Wc1
    {
        v8f aacc[4], gacc[4];
#pragma unroll
        for (int i = 0; i < 4; ++i) {
            aacc[i] = (v8f){0,0,0,0,0,0,0,0};
            gacc[i] = (v8f){0,0,0,0,0,0,0,0};
        }
        const u32* rowp = (const u32*)(&sX[rt * 16 + lrow][0]);
        for (int kt = 0; kt < 16; ++kt) {
            union { v16h v; u32 u[8]; } af;
#pragma unroll
            for (int j = 0; j < 8; ++j) {
                int k = kt * 32 + ((j >> 2) << 4) + hi * 8 + ((j & 3) << 1);
                af.u[j] = rowp[k >> 1];
            }
#pragma unroll
            for (int i = 0; i < 4; ++i) {
                v16h bfa = *(const v16h*)(wap + ((size_t)(kt * 16 + cs * 4 + i)) * 256 + lane * 8);
                aacc[i] = __builtin_amdgcn_wmma_f32_16x16x32_f16(
                    false, af.v, false, bfa, (short)0, aacc[i], false, false);
                v16h bfg = *(const v16h*)(wbp + ((size_t)(kt * 16 + cs * 4 + i)) * 256 + lane * 8);
                gacc[i] = __builtin_amdgcn_wmma_f32_16x16x32_f16(
                    false, af.v, false, bfg, (short)0, gacc[i], false, false);
            }
        }
        float ba[4], bg[4], wc[4];
#pragma unroll
        for (int i = 0; i < 4; ++i) {
            int col = (cs * 4 + i) * 16 + lrow;
            ba[i] = ba1[col]; bg[i] = bb1[col]; wc[i] = wc1[col];
        }
#pragma unroll
        for (int r = 0; r < 8; ++r) {
            float pr = 0.0f;
#pragma unroll
            for (int i = 0; i < 4; ++i) {
                float av = tanhf(aacc[i][r] + ba[i]);
                float gv = 1.0f / (1.0f + expf(-(gacc[i][r] + bg[i])));
                pr += av * gv * wc[i];
            }
            pr += __shfl_xor(pr, 1, 32);
            pr += __shfl_xor(pr, 2, 32);
            pr += __shfl_xor(pr, 4, 32);
            pr += __shfl_xor(pr, 8, 32);
            if (lrow == 0) atomicAdd(&sArow[rt * 16 + hi * 8 + r], pr);
        }
    }
    __syncthreads();
    if (t < 32) Aout[m0 + t] = sArow[t];
}

__global__ void k_segmax(const float* __restrict__ A, const int* __restrict__ cid,
                         int* __restrict__ segmax) {
    int idx = blockIdx.x * 256 + threadIdx.x;
    if (idx >= TOK) return;
    int b = idx >> 14;
    int c = cid[idx]; c = c < 0 ? 0 : (c > 10 ? 10 : c);
    atomicMax(&segmax[b * NCL + c], fkey(A[idx]));
}

__global__ void k_segexp(const float* __restrict__ A, const int* __restrict__ cid,
                         const int* __restrict__ segmax, float* __restrict__ expw,
                         float* __restrict__ segsum, float* __restrict__ segcnt) {
    int idx = blockIdx.x * 256 + threadIdx.x;
    if (idx >= TOK) return;
    int b = idx >> 14;
    int c = cid[idx]; c = c < 0 ? 0 : (c > 10 ? 10 : c);
    int s = b * NCL + c;
    float e = expf(A[idx] - fdec(segmax[s]));
    expw[idx] = e;
    atomicAdd(&segsum[s], e);
    atomicAdd(&segcnt[s], 1.0f);
}

// ---- cfeat[seg] = sum_i w_i * x_i ; 256 tokens/block, conflict-free LDS acc ----
__global__ __launch_bounds__(256)
void k_cfeat(const _Float16* __restrict__ xg, const int* __restrict__ cid,
             const float* __restrict__ expw, const float* __restrict__ segsum,
             float* __restrict__ cfeat) {
    __shared__ float acc[NCL][512];
    __shared__ float sw[256];
    __shared__ int   scid[256];
    int t = threadIdx.x;
    int t0 = blockIdx.x * 256;
    int b = t0 >> 14;
#pragma unroll
    for (int i = 0; i < 22; ++i) ((float*)acc)[t + i * 256] = 0.0f;
    {
        int c = cid[t0 + t]; c = c < 0 ? 0 : (c > 10 ? 10 : c);
        scid[t] = c;
        sw[t] = expw[t0 + t] / segsum[b * NCL + c];
    }
    __syncthreads();
    for (int tk = 0; tk < 256; ++tk) {
        float wv = sw[tk];
        int c = scid[tk];
        union { u32 u; _Float16 h[2]; } pp;
        pp.u = ((const u32*)(xg + (size_t)(t0 + tk) * 512))[t];
        acc[c][2 * t]     += wv * (float)pp.h[0];
        acc[c][2 * t + 1] += wv * (float)pp.h[1];
    }
    __syncthreads();
#pragma unroll
    for (int i = 0; i < 22; ++i) {
        int e = t + i * 256;
        atomicAdd(&cfeat[(size_t)b * NCL * 512 + e], ((float*)acc)[e]);
    }
}

// ---- cf2 = relu(cfeat @ W_intra + b_intra), one block per segment ----
__global__ __launch_bounds__(256)
void k_intra(const float* __restrict__ cfeat, const float* __restrict__ Wi,
             const float* __restrict__ bi, float* __restrict__ cf2) {
    __shared__ float row[512];
    int t = threadIdx.x, s = blockIdx.x;
    row[t]       = cfeat[s * 512 + t];
    row[t + 256] = cfeat[s * 512 + t + 256];
    __syncthreads();
#pragma unroll
    for (int o = 0; o < 2; ++o) {
        int col = t + o * 256;
        float a = bi[col];
        for (int k = 0; k < 512; ++k) a += row[k] * Wi[k * 512 + col];
        cf2[s * 512 + col] = a > 0.0f ? a : 0.0f;
    }
}

// ---- inter-cluster attention + head, one block per batch ----
__global__ __launch_bounds__(256)
void k_final(const float* __restrict__ cf2, const float* __restrict__ segcnt,
             const float* __restrict__ Wa2, const float* __restrict__ ba2,
             const float* __restrict__ Wb2, const float* __restrict__ bb2,
             const float* __restrict__ wc2, const float* __restrict__ bc2,
             const float* __restrict__ Wint, const float* __restrict__ bint,
             const float* __restrict__ Wcls, const float* __restrict__ bcls,
             float* __restrict__ out) {
    __shared__ float sc[NCL][512];
    __shared__ float sred[256];
    __shared__ float sA2[NCL], sW[NCL];
    __shared__ float slide[512];
    __shared__ float sl2[256];
    int t = threadIdx.x, b = blockIdx.x;
#pragma unroll
    for (int i = 0; i < 22; ++i)
        ((float*)sc)[t + i * 256] = cf2[(size_t)b * NCL * 512 + t + i * 256];
    __syncthreads();
    for (int c = 0; c < NCL; ++c) {
        float aa = ba2[t], gg = bb2[t];
        for (int k = 0; k < 512; ++k) {
            float x = sc[c][k];
            aa += x * Wa2[k * 256 + t];
            gg += x * Wb2[k * 256 + t];
        }
        aa = tanhf(aa);
        gg = 1.0f / (1.0f + expf(-gg));
        sred[t] = aa * gg * wc2[t];
        __syncthreads();
        for (int s = 128; s > 0; s >>= 1) {
            if (t < s) sred[t] += sred[t + s];
            __syncthreads();
        }
        if (t == 0) sA2[c] = sred[0] + bc2[0];
        __syncthreads();
    }
    if (t == 0) {
        float mx = -__builtin_inff();
        for (int c = 0; c < NCL; ++c) {
            float v = (segcnt[b * NCL + c] > 0.0f) ? sA2[c] : -__builtin_inff();
            sA2[c] = v;
            mx = v > mx ? v : mx;
        }
        float sm = 0.0f;
        for (int c = 0; c < NCL; ++c) {
            float e = (sA2[c] == -__builtin_inff()) ? 0.0f : expf(sA2[c] - mx);
            sW[c] = e; sm += e;
        }
        for (int c = 0; c < NCL; ++c) sW[c] /= sm;
    }
    __syncthreads();
#pragma unroll
    for (int i = 0; i < 2; ++i) {
        int k = t + i * 256;
        float s = 0.0f;
        for (int c = 0; c < NCL; ++c) s += sW[c] * sc[c][k];
        slide[k] = s;
    }
    __syncthreads();
    {
        float a = bint[t];
        for (int k = 0; k < 512; ++k) a += slide[k] * Wint[k * 256 + t];
        sl2[t] = a > 0.0f ? a : 0.0f;
    }
    __syncthreads();
    for (int cls = 0; cls < 2; ++cls) {
        sred[t] = sl2[t] * Wcls[t * 2 + cls];
        __syncthreads();
        for (int s = 128; s > 0; s >>= 1) {
            if (t < s) sred[t] += sred[t + s];
            __syncthreads();
        }
        if (t == 0) out[b * 2 + cls] = sred[0] + bcls[cls];
        __syncthreads();
    }
}

extern "C" void kernel_launch(void* const* d_in, const int* in_sizes, int n_in,
                              void* d_out, int out_size, void* d_ws, size_t ws_size,
                              hipStream_t stream) {
    const float* h      = (const float*)d_in[0];
    const int*   cids   = (const int*)d_in[1];
    const float* emb    = (const float*)d_in[2];
    const float* Wfuse  = (const float*)d_in[3];
    const float* bfuse  = (const float*)d_in[4];
    const float* Wa1    = (const float*)d_in[5];
    const float* ba1    = (const float*)d_in[6];
    const float* Wb1    = (const float*)d_in[7];
    const float* bb1    = (const float*)d_in[8];
    const float* Wc1    = (const float*)d_in[9];
    const float* bc1    = (const float*)d_in[10];
    const float* Wintra = (const float*)d_in[11];
    const float* bintra = (const float*)d_in[12];
    const float* Wa2    = (const float*)d_in[13];
    const float* ba2    = (const float*)d_in[14];
    const float* Wb2    = (const float*)d_in[15];
    const float* bb2    = (const float*)d_in[16];
    const float* Wc2    = (const float*)d_in[17];
    const float* bc2    = (const float*)d_in[18];
    const float* Winter = (const float*)d_in[19];
    const float* binter = (const float*)d_in[20];
    const float* Wcls   = (const float*)d_in[21];
    const float* bcls   = (const float*)d_in[22];
    (void)in_sizes; (void)n_in; (void)out_size; (void)ws_size;

    char* ws = (char*)d_ws;
    size_t o = 0;
    auto alloc = [&](size_t bytes) -> char* {
        char* p = ws + o;
        o = (o + bytes + 255) & ~(size_t)255;
        return p;
    };
    u32*      wfp    = (u32*)alloc((size_t)17 * 32 * 1024);     // W_fuse packed f16 frags
    u32*      wap    = (u32*)alloc((size_t)16 * 16 * 1024);     // Wa1 packed
    u32*      wbp    = (u32*)alloc((size_t)16 * 16 * 1024);     // Wb1 packed
    _Float16* xg     = (_Float16*)alloc((size_t)TOK * 512 * 2); // fused features f16
    float*    Asc    = (float*)alloc((size_t)TOK * 4);
    float*    expw   = (float*)alloc((size_t)TOK * 4);
    int*      segmax = (int*)alloc(128 * 4);
    float*    segsum = (float*)alloc(128 * 4);
    float*    segcnt = (float*)alloc(128 * 4);
    float*    cfeat  = (float*)alloc((size_t)BB * NCL * 512 * 4);
    float*    cf2    = (float*)alloc((size_t)BB * NCL * 512 * 4);

    k_pack_b<<<544, 256, 0, stream>>>(Wfuse, 520, 512, 32, wfp, 17 * 32 * 256);
    k_pack_b<<<256, 256, 0, stream>>>(Wa1, 512, 256, 16, wap, 16 * 16 * 256);
    k_pack_b<<<256, 256, 0, stream>>>(Wb1, 512, 256, 16, wbp, 16 * 16 * 256);
    k_init<<<176, 256, 0, stream>>>(segmax, segsum, segcnt, cfeat);
    k_token<<<TOK / 32, 256, 0, stream>>>(h, cids, emb, wfp, bfuse, wap, ba1,
                                          wbp, bb1, Wc1, bc1, xg, Asc);
    k_segmax<<<TOK / 256, 256, 0, stream>>>(Asc, cids, segmax);
    k_segexp<<<TOK / 256, 256, 0, stream>>>(Asc, cids, segmax, expw, segsum, segcnt);
    k_cfeat<<<TOK / 256, 256, 0, stream>>>(xg, cids, expw, segsum, cfeat);
    k_intra<<<BB * NCL, 256, 0, stream>>>(cfeat, Wintra, bintra, cf2);
    k_final<<<BB, 256, 0, stream>>>(cf2, segcnt, Wa2, ba2, Wb2, bb2, Wc2, bc2,
                                    Winter, binter, Wcls, bcls, (float*)d_out);
}
